// GNNEncoder_61323543052635
// MI455X (gfx1250) — matrix operand order repaired and checked by reading, not compiled, not run on hardware
//
#include <hip/hip_runtime.h>
#include <hip/hip_bf16.h>

typedef __attribute__((ext_vector_type(16))) __bf16 v16bf;
typedef __attribute__((ext_vector_type(8)))  float  v8f;

union ABFrag { v16bf v; __bf16 e[16]; };

#define Bn   16384
#define Dn   256
#define Hn   256
#define ASTR 264   // padded bf16 LDS row stride (264*2B = 33 uint4, staggers banks)
#define DSTR 65    // padded f32 LDS row stride for distance tile

// knn LDS layout (bytes)
#define KNN_A_BYTES   67584            // 128*ASTR*2  (one of Ah/Al)
#define KNN_BBUF      33792            // 64*ASTR*2   (one of Bh/Bl)
#define KNN_BPAIR     67584            // Bh+Bl for one buffer
#define KNN_BOFF      135168           // after Ah+Al
#define KNN_DTOFF     270336           // after 2 double-buffered B pairs
#define KNN_SQAOFF    303616
#define KNN_LDS_TOTAL 304128

// ---------------------------------------------------------------- prep_x
// one wave per row: sq-norm + bf16 hi/lo split of x
__global__ __launch_bounds__(256) void gnn_prep_x(
    const float* __restrict__ x, __bf16* __restrict__ xhi,
    __bf16* __restrict__ xlo, float* __restrict__ sq)
{
  const int tid  = threadIdx.x;
  const int lane = tid & 31;
  const int row  = blockIdx.x * 8 + (tid >> 5);
  const float* xr = x + (size_t)row * Dn;
  float s = 0.f;
#pragma unroll
  for (int c = 0; c < 2; ++c) {
    float4 f = ((const float4*)xr)[lane * 2 + c];
    const int base = (lane * 2 + c) * 4;
    float e[4] = {f.x, f.y, f.z, f.w};
#pragma unroll
    for (int i = 0; i < 4; ++i) {
      float v = e[i];
      s += v * v;
      __bf16 h = (__bf16)v;
      xhi[(size_t)row * Dn + base + i] = h;
      xlo[(size_t)row * Dn + base + i] = (__bf16)(v - (float)h);
    }
  }
#pragma unroll
  for (int off = 16; off > 0; off >>= 1) s += __shfl_xor(s, off, 32);
  if (lane == 0) sq[row] = s;
}

// ---------------------------------------------------------------- prep_w
__global__ __launch_bounds__(256) void gnn_prep_w(
    const float* __restrict__ w1, const float* __restrict__ w2,
    __bf16* __restrict__ w1hi, __bf16* __restrict__ w1lo,
    __bf16* __restrict__ w2hi, __bf16* __restrict__ w2lo)
{
  const int i = blockIdx.x * 256 + threadIdx.x;  // 65536 total
  float a = w1[i]; __bf16 ha = (__bf16)a;
  w1hi[i] = ha; w1lo[i] = (__bf16)(a - (float)ha);
  float b = w2[i]; __bf16 hb = (__bf16)b;
  w2hi[i] = hb; w2lo[i] = (__bf16)(b - (float)hb);
}

// ---------------------------------------------------------------- knn
// WG = 128 threads (4 waves) owns 128 rows; streams 64-col tiles of the
// Gram via split-bf16 WMMA with async double-buffered B staging (compile-
// time-constant buffer addresses, 2 accumulators in flight); keeps a
// register-resident sorted top-5 per row.
__global__ __launch_bounds__(128) void gnn_knn(
    const __bf16* __restrict__ xhi, const __bf16* __restrict__ xlo,
    const float* __restrict__ sqn, int* __restrict__ knn_out)
{
  extern __shared__ char smem[];
  __bf16* Ah  = (__bf16*)smem;                              // [128][ASTR]
  __bf16* Al  = (__bf16*)(smem + KNN_A_BYTES);              // [128][ASTR]
  const __bf16* Bh0 = (const __bf16*)(smem + KNN_BOFF);                           // [64][ASTR]
  const __bf16* Bl0 = (const __bf16*)(smem + KNN_BOFF + KNN_BBUF);
  const __bf16* Bh1 = (const __bf16*)(smem + KNN_BOFF + KNN_BPAIR);
  const __bf16* Bl1 = (const __bf16*)(smem + KNN_BOFF + KNN_BPAIR + KNN_BBUF);
  float*  Dt  = (float*)(smem + KNN_DTOFF);                 // [128][DSTR]
  float*  sqA = (float*)(smem + KNN_SQAOFF);                // [128]

  const int tid     = threadIdx.x;
  const int wave    = tid >> 5;
  const int lane    = tid & 31;
  const int rowBase = blockIdx.x * 128;

  // LDS byte offset of the B-buffer region (generic LDS ptr: low 32 bits = LDS addr)
  const unsigned ldsB0 = (unsigned)(uintptr_t)(smem + KNN_BOFF);

  // resident A block (hi+lo), padded rows (regular path; done once)
  for (int c = tid; c < 128 * 32; c += 128) {
    const int r = c >> 5, p = c & 31;
    ((uint4*)Ah)[r * 33 + p] = ((const uint4*)(xhi + (size_t)(rowBase + r) * Dn))[p];
    ((uint4*)Al)[r * 33 + p] = ((const uint4*)(xlo + (size_t)(rowBase + r) * Dn))[p];
  }
  sqA[tid] = sqn[rowBase + tid];

  // async-issue one 64-row B tile (hi+lo) into LDS buffer at byte offset bofs.
  // 16 loop iters/thread x 2 ops = 32 async instructions per wave per tile.
#define ISSUE_TILE(cb_, bofs_)                                                  \
  {                                                                             \
    const unsigned lb = ldsB0 + (unsigned)(bofs_);                              \
    for (int c = tid; c < 64 * 32; c += 128) {                                  \
      const int r = c >> 5, p = c & 31;                                         \
      const char* gh = (const char*)(xhi + (size_t)((cb_) + r) * Dn) + p * 16;  \
      const char* gl = (const char*)(xlo + (size_t)((cb_) + r) * Dn) + p * 16;  \
      const unsigned lo_ = lb + (unsigned)(r * 33 + p) * 16;                    \
      asm volatile("global_load_async_to_lds_b128 %0, %1, off"                  \
                   :: "v"(lo_), "v"(gh) : "memory");                            \
      asm volatile("global_load_async_to_lds_b128 %0, %1, off"                  \
                   :: "v"(lo_ + (unsigned)KNN_BBUF), "v"(gl) : "memory");       \
    }                                                                           \
  }

  float bd0 = 3.4e38f, bd1 = 3.4e38f, bd2 = 3.4e38f, bd3 = 3.4e38f, bd4 = 3.4e38f;
  int   bi0 = 0, bi1 = 0, bi2 = 0, bi3 = 0, bi4 = 0;

  const int m16   = lane & 15;
  const int lh    = lane >> 4;
  const int akoff = lh * 8;    // A frag: K chunks [akoff..+8) and [akoff+16..+24)
  const int bkoff = lh * 8;    // B frag: K chunks [bkoff..+8) and [bkoff+16..+24)
  const int mhalf = lh * 8;    // C/D: VGPR v -> row mhalf+v

  // GEMM + distance epilogue for one 128x64 tile from a (compile-time) buffer.
  // 2 accumulators in flight -> independent WMMAs fill co-exec hazard slots.
  auto compute_tile = [&](const __bf16* Bh, const __bf16* Bl, int cb) {
    for (int mt = 0; mt < 2; ++mt) {
      const int mrow = wave * 32 + mt * 16;
      const __bf16* arh = &Ah[(size_t)(mrow + m16) * ASTR + akoff];
      const __bf16* arl = &Al[(size_t)(mrow + m16) * ASTR + akoff];
      for (int np = 0; np < 2; ++np) {
        const int nt0 = np * 2, nt1 = np * 2 + 1;
        const __bf16* brh0 = &Bh[(size_t)(nt0 * 16 + m16) * ASTR + bkoff];
        const __bf16* brl0 = &Bl[(size_t)(nt0 * 16 + m16) * ASTR + bkoff];
        const __bf16* brh1 = &Bh[(size_t)(nt1 * 16 + m16) * ASTR + bkoff];
        const __bf16* brl1 = &Bl[(size_t)(nt1 * 16 + m16) * ASTR + bkoff];
        v8f acc0 = {0.f, 0.f, 0.f, 0.f, 0.f, 0.f, 0.f, 0.f};
        v8f acc1 = {0.f, 0.f, 0.f, 0.f, 0.f, 0.f, 0.f, 0.f};
#pragma unroll
        for (int kk = 0; kk < Dn; kk += 32) {
          ABFrag ah, al, bh0, bl0, bh1, bl1;
#pragma unroll
          for (int i = 0; i < 8; ++i) {
            ah.e[i] = arh[kk + i];      ah.e[8 + i] = arh[kk + 16 + i];
            al.e[i] = arl[kk + i];      al.e[8 + i] = arl[kk + 16 + i];
          }
#pragma unroll
          for (int i = 0; i < 8; ++i) {
            bh0.e[i] = brh0[kk + i];    bh0.e[8 + i] = brh0[kk + 16 + i];
            bl0.e[i] = brl0[kk + i];    bl0.e[8 + i] = brl0[kk + 16 + i];
            bh1.e[i] = brh1[kk + i];    bh1.e[8 + i] = brh1[kk + 16 + i];
            bl1.e[i] = brl1[kk + i];    bl1.e[8 + i] = brl1[kk + 16 + i];
          }
          acc0 = __builtin_amdgcn_wmma_f32_16x16x32_bf16(false, ah.v, false, bh0.v, (short)0, acc0, false, false);
          acc1 = __builtin_amdgcn_wmma_f32_16x16x32_bf16(false, ah.v, false, bh1.v, (short)0, acc1, false, false);
          acc0 = __builtin_amdgcn_wmma_f32_16x16x32_bf16(false, ah.v, false, bl0.v, (short)0, acc0, false, false);
          acc1 = __builtin_amdgcn_wmma_f32_16x16x32_bf16(false, ah.v, false, bl1.v, (short)0, acc1, false, false);
          acc0 = __builtin_amdgcn_wmma_f32_16x16x32_bf16(false, al.v, false, bh0.v, (short)0, acc0, false, false);
          acc1 = __builtin_amdgcn_wmma_f32_16x16x32_bf16(false, al.v, false, bh1.v, (short)0, acc1, false, false);
        }
        const float sb0 = sqn[cb + nt0 * 16 + m16];  // L2-resident
        const float sb1 = sqn[cb + nt1 * 16 + m16];
#pragma unroll
        for (int v = 0; v < 8; ++v) {
          const int m = mhalf + v;
          const float d20 = sqA[mrow + m] + sb0 - 2.0f * acc0[v];
          const float d21 = sqA[mrow + m] + sb1 - 2.0f * acc1[v];
          Dt[(size_t)(mrow + m) * DSTR + nt0 * 16 + m16] = fmaxf(d20, 0.0f);
          Dt[(size_t)(mrow + m) * DSTR + nt1 * 16 + m16] = fmaxf(d21, 0.0f);
        }
      }
    }
  };

  // per-row top-5 update (thread tid owns row rowBase+tid)
  auto scan_tile = [&](int cb) {
    const float* drow = &Dt[(size_t)tid * DSTR];
    for (int c = 0; c < 64; ++c) {
      const float d = drow[c];
      if (d < bd4) {
        const int j = cb + c;
        bd4 = d; bi4 = j;
        if (bd4 < bd3) { float t = bd3; bd3 = bd4; bd4 = t; int ti = bi3; bi3 = bi4; bi4 = ti; }
        if (bd3 < bd2) { float t = bd2; bd2 = bd3; bd3 = t; int ti = bi2; bi2 = bi3; bi3 = ti; }
        if (bd2 < bd1) { float t = bd1; bd1 = bd2; bd2 = t; int ti = bi1; bi1 = bi2; bi2 = ti; }
        if (bd1 < bd0) { float t = bd0; bd0 = bd1; bd1 = t; int ti = bi0; bi0 = bi1; bi1 = ti; }
      }
    }
  };

  // prologue: tile 0 -> buffer 0
  ISSUE_TILE(0, 0)

  const int nTiles = Bn / 64;  // 256, even
  for (int it = 0; it < nTiles; it += 2) {
    // ---- half 0: consume buffer 0 (tile it), prefetch tile it+1 -> buffer 1
    ISSUE_TILE((it + 1) * 64, KNN_BPAIR)
    asm volatile("s_wait_asynccnt 0x20" ::: "memory");  // tile `it` complete
    __syncthreads();
    compute_tile(Bh0, Bl0, it * 64);
    __syncthreads();
    scan_tile(it * 64);

    // ---- half 1: consume buffer 1 (tile it+1), prefetch tile it+2 -> buffer 0
    if (it + 2 < nTiles) {
      ISSUE_TILE((it + 2) * 64, 0)
      asm volatile("s_wait_asynccnt 0x20" ::: "memory");  // tile `it+1` complete
    } else {
      asm volatile("s_wait_asynccnt 0x0" ::: "memory");
    }
    __syncthreads();
    compute_tile(Bh1, Bl1, (it + 1) * 64);
    __syncthreads();
    scan_tile((it + 1) * 64);
  }
#undef ISSUE_TILE
  int* ko = knn_out + (size_t)(rowBase + tid) * 5;
  ko[0] = bi0; ko[1] = bi1; ko[2] = bi2; ko[3] = bi3; ko[4] = bi4;
}

// ---------------------------------------------------------------- mlp1
// WG = 128 threads owns 16 b-rows -> 80 neighbor rows; h=relu(nb@W1^T+b1),
// mean over k -> agg[16][256]
__global__ __launch_bounds__(128) void gnn_mlp1(
    const __bf16* __restrict__ xhi, const __bf16* __restrict__ xlo,
    const int* __restrict__ knn_in,
    const __bf16* __restrict__ w1hi, const __bf16* __restrict__ w1lo,
    const float* __restrict__ b1, float* __restrict__ agg)
{
  extern __shared__ char smem[];
  __bf16* Nh  = (__bf16*)smem;             // [80][ASTR]
  __bf16* Nl  = Nh + 80 * ASTR;
  float*  Hls = (float*)(Nl + 80 * ASTR);  // [80][256]

  const int tid   = threadIdx.x;
  const int wave  = tid >> 5;
  const int lane  = tid & 31;
  const int bBase = blockIdx.x * 16;

  for (int c = tid; c < 80 * 32; c += 128) {
    const int r = c >> 5, p = c & 31;
    const int src = knn_in[(size_t)(bBase + r / 5) * 5 + (r % 5)];
    ((uint4*)Nh)[r * 33 + p] = ((const uint4*)(xhi + (size_t)src * Dn))[p];
    ((uint4*)Nl)[r * 33 + p] = ((const uint4*)(xlo + (size_t)src * Dn))[p];
  }
  __syncthreads();

  const int m16 = lane & 15, lh = lane >> 4;
  const int akoff = lh * 8, bkoff = lh * 8, mhalf = lh * 8;

  for (int mt = 0; mt < 5; ++mt) {
    const __bf16* arh = &Nh[(size_t)(mt * 16 + m16) * ASTR + akoff];
    const __bf16* arl = &Nl[(size_t)(mt * 16 + m16) * ASTR + akoff];
    for (int t = 0; t < 4; ++t) {
      const int nt = wave * 4 + t;
      const __bf16* brh = w1hi + (size_t)(nt * 16 + m16) * Dn + bkoff;
      const __bf16* brl = w1lo + (size_t)(nt * 16 + m16) * Dn + bkoff;
      v8f acc = {0.f, 0.f, 0.f, 0.f, 0.f, 0.f, 0.f, 0.f};
#pragma unroll
      for (int kk = 0; kk < Dn; kk += 32) {
        ABFrag ah, al, bh, bl;
#pragma unroll
        for (int i = 0; i < 8; ++i) {
          ah.e[i] = arh[kk + i];      ah.e[8 + i] = arh[kk + 16 + i];
          al.e[i] = arl[kk + i];      al.e[8 + i] = arl[kk + 16 + i];
        }
#pragma unroll
        for (int i = 0; i < 8; ++i) { bh.e[i] = brh[kk + i]; bh.e[8 + i] = brh[kk + 16 + i]; bl.e[i] = brl[kk + i]; bl.e[8 + i] = brl[kk + 16 + i]; }
        acc = __builtin_amdgcn_wmma_f32_16x16x32_bf16(false, ah.v, false, bh.v, (short)0, acc, false, false);
        acc = __builtin_amdgcn_wmma_f32_16x16x32_bf16(false, ah.v, false, bl.v, (short)0, acc, false, false);
        acc = __builtin_amdgcn_wmma_f32_16x16x32_bf16(false, al.v, false, bh.v, (short)0, acc, false, false);
      }
      const float bias = b1[nt * 16 + m16];
#pragma unroll
      for (int v = 0; v < 8; ++v) {
        const int m = mhalf + v;
        Hls[(size_t)(mt * 16 + m) * Hn + nt * 16 + m16] = fmaxf(acc[v] + bias, 0.f);
      }
    }
  }
  __syncthreads();

  for (int o = tid; o < 16 * Hn; o += 128) {
    const int b = o >> 8, hh = o & 255;
    float s = 0.f;
#pragma unroll
    for (int q = 0; q < 5; ++q) s += Hls[(size_t)(b * 5 + q) * Hn + hh];
    agg[(size_t)(bBase + b) * Hn + hh] = s * 0.2f;
  }
}

// ---------------------------------------------------------------- mlp2
// out = agg @ W2^T + b2 ; WG = 128 threads owns 64 rows
__global__ __launch_bounds__(128) void gnn_mlp2(
    const float* __restrict__ agg,
    const __bf16* __restrict__ w2hi, const __bf16* __restrict__ w2lo,
    const float* __restrict__ b2, float* __restrict__ out)
{
  extern __shared__ char smem[];
  __bf16* Ah = (__bf16*)smem;       // [64][ASTR]
  __bf16* Al = Ah + 64 * ASTR;

  const int tid   = threadIdx.x;
  const int wave  = tid >> 5;
  const int lane  = tid & 31;
  const int rBase = blockIdx.x * 64;

  for (int i = tid; i < 64 * Hn; i += 128) {
    const int r = i >> 8, c = i & 255;
    const float f = agg[(size_t)(rBase + r) * Hn + c];
    __bf16 h = (__bf16)f;
    Ah[(size_t)r * ASTR + c] = h;
    Al[(size_t)r * ASTR + c] = (__bf16)(f - (float)h);
  }
  __syncthreads();

  const int m16 = lane & 15, lh = lane >> 4;
  const int akoff = lh * 8, bkoff = lh * 8, mhalf = lh * 8;

  for (int mt = 0; mt < 4; ++mt) {
    const __bf16* arh = &Ah[(size_t)(mt * 16 + m16) * ASTR + akoff];
    const __bf16* arl = &Al[(size_t)(mt * 16 + m16) * ASTR + akoff];
    for (int t = 0; t < 4; ++t) {
      const int nt = wave * 4 + t;
      const __bf16* brh = w2hi + (size_t)(nt * 16 + m16) * Hn + bkoff;
      const __bf16* brl = w2lo + (size_t)(nt * 16 + m16) * Hn + bkoff;
      v8f acc = {0.f, 0.f, 0.f, 0.f, 0.f, 0.f, 0.f, 0.f};
#pragma unroll
      for (int kk = 0; kk < Hn; kk += 32) {
        ABFrag ah, al, bh, bl;
#pragma unroll
        for (int i = 0; i < 8; ++i) {
          ah.e[i] = arh[kk + i];      ah.e[8 + i] = arh[kk + 16 + i];
          al.e[i] = arl[kk + i];      al.e[8 + i] = arl[kk + 16 + i];
        }
#pragma unroll
        for (int i = 0; i < 8; ++i) { bh.e[i] = brh[kk + i]; bh.e[8 + i] = brh[kk + 16 + i]; bl.e[i] = brl[kk + i]; bl.e[8 + i] = brl[kk + 16 + i]; }
        acc = __builtin_amdgcn_wmma_f32_16x16x32_bf16(false, ah.v, false, bh.v, (short)0, acc, false, false);
        acc = __builtin_amdgcn_wmma_f32_16x16x32_bf16(false, ah.v, false, bl.v, (short)0, acc, false, false);
        acc = __builtin_amdgcn_wmma_f32_16x16x32_bf16(false, al.v, false, bh.v, (short)0, acc, false, false);
      }
      const float bias = b2[nt * 16 + m16];
#pragma unroll
      for (int v = 0; v < 8; ++v) {
        const int m = mhalf + v;
        out[(size_t)(rBase + mt * 16 + m) * Hn + nt * 16 + m16] = acc[v] + bias;
      }
    }
  }
}

// ---------------------------------------------------------------- launch
extern "C" void kernel_launch(void* const* d_in, const int* in_sizes, int n_in,
                              void* d_out, int out_size, void* d_ws, size_t ws_size,
                              hipStream_t stream) {
  (void)in_sizes; (void)n_in; (void)out_size; (void)ws_size;
  const float* x  = (const float*)d_in[0];
  const float* W1 = (const float*)d_in[1];
  const float* b1 = (const float*)d_in[2];
  const float* W2 = (const float*)d_in[3];
  const float* b2 = (const float*)d_in[4];
  float* out = (float*)d_out;
  char* ws = (char*)d_ws;

  __bf16* xhi  = (__bf16*)(ws);                 //  8,388,608 B
  __bf16* xlo  = (__bf16*)(ws + 8388608);       //  8,388,608 B
  __bf16* w1hi = (__bf16*)(ws + 16777216);      //    131,072 B
  __bf16* w1lo = (__bf16*)(ws + 16908288);
  __bf16* w2hi = (__bf16*)(ws + 17039360);
  __bf16* w2lo = (__bf16*)(ws + 17170432);
  float*  sq   = (float*)(ws + 17301504);       //     65,536 B
  int*    knn  = (int*)(ws + 17367040);         //    327,680 B
  float*  agg  = (float*)(ws + 17694720);       // 16,777,216 B  (total ~34.5 MB)

  gnn_prep_x<<<Bn / 8, 256, 0, stream>>>(x, xhi, xlo, sq);
  gnn_prep_w<<<65536 / 256, 256, 0, stream>>>(W1, W2, w1hi, w1lo, w2hi, w2lo);

  const size_t lds_knn  = KNN_LDS_TOTAL;                                 // 304,128 B
  const size_t lds_mlp1 = (size_t)(2 * 80 * ASTR) * 2 + (size_t)(80 * Hn) * 4; // 166,400 B
  const size_t lds_mlp2 = (size_t)(2 * 64 * ASTR) * 2;                   //  67,584 B

  gnn_knn <<<Bn / 128, 128, lds_knn,  stream>>>(xhi, xlo, sq, knn);
  gnn_mlp1<<<Bn / 16,  128, lds_mlp1, stream>>>(xhi, xlo, knn, w1hi, w1lo, b1, agg);
  gnn_mlp2<<<Bn / 64,  128, lds_mlp2, stream>>>(agg, w2hi, w2lo, b2, out);
}
